// Meta_39444979647206
// MI455X (gfx1250) — matrix operand order repaired and checked by reading, not compiled, not run on hardware
//
#include <hip/hip_runtime.h>

// ---------------------------------------------------------------------------
// GNN (MetaLayer) forward for MI455X / gfx1250.
// Matmuls via V_WMMA_F32_16X16X32_BF16 (bf16 operands, fp32 accumulation).
// Weights are pre-packed once per call into bf16 *in WMMA B-fragment order*,
// so each WMMA's B operand is one coalesced 32-byte load; A-fragments are
// converted from LDS once per layer and reused across all 4 n-tiles.
// ---------------------------------------------------------------------------

constexpr int NUM_GRAPHS = 256;
constexpr int N_NODES    = 100000;
constexpr int N_EDGES    = 500000;
constexpr int H64        = 64;

typedef __attribute__((ext_vector_type(8)))  float  v8f;
typedef __attribute__((ext_vector_type(16))) __bf16 v16bf;

__device__ __forceinline__ void cfence() { asm volatile("" ::: "memory"); }

struct Seg  { const float* ptr; const int* idx; int width; };
// packed bf16 weights (B-frag layout) + f32 bias/LN vectors
struct MLPW { const __bf16 *W0,*W1,*W2,*W3;
              const float  *b0,*b1,*b2,*g,*bb,*b3; };
struct DecHead { const float *g0,*b0,*W0,*bb0,*g1,*b1,*W1,*bb1; };

enum { OUT_STORE = 0, OUT_ADD = 1, OUT_SCATTER = 2 };

// Load all K/32 A-fragments for this wave's 16-row tile from LDS (f32).
// A (16x32, 16-bit): lane = row ln; elem e -> K = k0 + 16*half + e
template <int K>
__device__ __forceinline__ void load_afrags(v16bf* afr, const float* Xs, int ldx,
                                            int ln, int half) {
  #pragma unroll
  for (int kb = 0; kb < K / 32; ++kb) {
    #pragma unroll
    for (int e = 0; e < 16; ++e) {
      const int ka = kb * 32 + 16 * half + e;
      afr[kb][e] = (__bf16)Xs[ln * ldx + ka];
    }
  }
}

// Hidden layer: Y(16x64) = act(X(16xK) @ W(Kx64) + b).
// Wp layout: [kb][nt][lane][16] bf16 -> one v16bf (32B) load per WMMA.
template <int K>
__device__ __forceinline__ void layer(const float* Xs, int ldx,
                                      const __bf16* __restrict__ Wp,
                                      const float* __restrict__ bias,
                                      float* Ys, int ldy, bool relu,
                                      int ln, int half, int lane) {
  v16bf afr[K / 32];
  load_afrags<K>(afr, Xs, ldx, ln, half);
  #pragma unroll
  for (int nt = 0; nt < 4; ++nt) {
    v8f acc = {0.f, 0.f, 0.f, 0.f, 0.f, 0.f, 0.f, 0.f};
    #pragma unroll
    for (int kb = 0; kb < K / 32; ++kb) {
      const v16bf b = *(const v16bf*)(Wp + ((size_t)((kb * 4 + nt) * 32 + lane)) * 16);
      acc = __builtin_amdgcn_wmma_f32_16x16x32_bf16(false, afr[kb], false, b,
                                                    (short)0, acc, false, false);
    }
    const float bv = bias[nt * 16 + ln];
    #pragma unroll
    for (int v = 0; v < 8; ++v) {
      float val = acc[v] + bv;
      if (relu) val = fmaxf(val, 0.f);
      Ys[(v + half * 8) * ldy + nt * 16 + ln] = val;   // D: row = v+8*half, col = ln
    }
  }
  cfence();
}

__device__ __forceinline__ void layernorm(float* Ys, int ldy,
                                          const float* __restrict__ g,
                                          const float* __restrict__ b,
                                          int ln, int half) {
  if (half == 0) {  // lanes 0..15: one row each
    float m = 0.f;
    #pragma unroll
    for (int c = 0; c < 64; ++c) m += Ys[ln * ldy + c];
    m *= (1.f / 64.f);
    float vv = 0.f;
    #pragma unroll
    for (int c = 0; c < 64; ++c) { float d = Ys[ln * ldy + c] - m; vv += d * d; }
    const float rs = rsqrtf(vv * (1.f / 64.f) + 1e-5f);
    #pragma unroll
    for (int c = 0; c < 64; ++c) {
      const float d = Ys[ln * ldy + c];
      Ys[ln * ldy + c] = (d - m) * rs * g[c] + b[c];
    }
  }
  cfence();
}

// Fused 4-layer MLP (relu,relu,relu,LN,linear). 4 waves/block, 16 rows/wave.
template <int KIN>
__global__ void __launch_bounds__(128)
fused_mlp(MLPW w, Seg s0, Seg s1, Seg s2, Seg s3, Seg s4, int nseg,
          float* __restrict__ out, const int* __restrict__ out_idx,
          int mode, int ntiles) {
  extern __shared__ float smem[];
  const int lane = threadIdx.x & 31;
  const int wid  = threadIdx.x >> 5;
  const int half = lane >> 4;
  const int ln   = lane & 15;
  const int LDA  = KIN + 4;                      // pad -> bank-conflict-free
  const int perWave = 16 * LDA + 16 * 68;
  float* bufA = smem + wid * perWave;
  float* bufB = bufA + 16 * LDA;

  const int tile = blockIdx.x * 4 + wid;
  if (tile >= ntiles) return;                    // wave-uniform
  const int tb = tile * 16;

  Seg segs[5] = {s0, s1, s2, s3, s4};

  // ---- stage gathered/concatenated input tile (16 x KIN) into LDS with
  //      float4 (b128) gathers; uncovered columns are zero-filled. ----
  constexpr int KD4 = KIN / 4;
  for (int i = lane; i < 16 * KD4; i += 32) {
    const int r = i / KD4, c = (i - r * KD4) * 4;
    int off = c;
    float4 val = make_float4(0.f, 0.f, 0.f, 0.f);
    #pragma unroll
    for (int q = 0; q < 5; ++q) {
      if (q < nseg) {
        const int wd = segs[q].width;            // always a multiple of 4
        if (off < wd) {
          const int rr = segs[q].idx ? segs[q].idx[tb + r] : (tb + r);
          val = *(const float4*)(segs[q].ptr + (size_t)rr * wd + off);
          off = 1 << 30;
        } else off -= wd;
      }
    }
    *(float4*)(bufA + r * LDA + c) = val;
  }
  cfence();

  layer<KIN>(bufA, LDA, w.W0, w.b0, bufB, 68, true, ln, half, lane);
  layer<64>(bufB, 68, w.W1, w.b1, bufA, 68, true, ln, half, lane);
  layer<64>(bufA, 68, w.W2, w.b2, bufB, 68, true, ln, half, lane);
  layernorm(bufB, 68, w.g, w.bb, ln, half);

  // ---- output layer (linear) straight to global ----
  {
    v16bf afr[2];
    load_afrags<64>(afr, bufB, 68, ln, half);
    for (int nt = 0; nt < 4; ++nt) {
      v8f acc = {0.f, 0.f, 0.f, 0.f, 0.f, 0.f, 0.f, 0.f};
      #pragma unroll
      for (int kb = 0; kb < 2; ++kb) {
        const v16bf b = *(const v16bf*)(w.W3 + ((size_t)((kb * 4 + nt) * 32 + lane)) * 16);
        acc = __builtin_amdgcn_wmma_f32_16x16x32_bf16(false, afr[kb], false, b,
                                                      (short)0, acc, false, false);
      }
      const float bv = w.b3[nt * 16 + ln];
      #pragma unroll
      for (int v = 0; v < 8; ++v) {
        const float val = acc[v] + bv;
        const int rl = v + half * 8;
        size_t orow = (mode == OUT_SCATTER) ? (size_t)out_idx[tb + rl]
                                            : (size_t)(tb + rl);
        float* p = out + orow * H64 + nt * 16 + ln;
        if (mode == OUT_SCATTER)      atomicAdd(p, val);
        else if (mode == OUT_ADD)     *p += val;
        else                          *p = val;
      }
    }
  }
}

// ---------------------------- helper kernels -------------------------------

// Pre-pack f32 weights (K x 64 row-major, rows >= Ksrc treated as zero) into
// bf16 in WMMA B-fragment order: dst[((kb*4+nt)*32+lane)*16+e] = W[k][n]
// with k = kb*32 + 16*(lane>>4) + e, n = nt*16 + (lane&15).
struct RepackEnt { const float* src; __bf16* dst; int K; int Ksrc; };
struct RepackTab { RepackEnt e[56]; };

__global__ void k_repack(RepackTab tab) {
  const RepackEnt en = tab.e[blockIdx.x];
  const int total = en.K * 64;
  for (int d = threadIdx.x; d < total; d += blockDim.x) {
    const int e    = d & 15;
    const int lane = (d >> 4) & 31;
    const int nt   = (d >> 9) & 3;
    const int kb   = d >> 11;
    const int k = kb * 32 + 16 * (lane >> 4) + e;
    const int n = nt * 16 + (lane & 15);
    const float v = (k < en.Ksrc) ? en.src[k * 64 + n] : 0.f;
    en.dst[d] = (__bf16)v;
  }
}

__global__ void k_zero(float* p, long long n) {
  long long i = (long long)blockIdx.x * blockDim.x + threadIdx.x;
  if (i < n) p[i] = 0.f;
}

__global__ void k_eb(const int* __restrict__ row, const int* __restrict__ batch,
                     int* __restrict__ eb, int E) {
  int i = blockIdx.x * blockDim.x + threadIdx.x;
  if (i < E) eb[i] = batch[row[i]];
}

__global__ void k_eenc(const float* __restrict__ x, const float* __restrict__ ea,
                       const int* __restrict__ row, const int* __restrict__ col,
                       float* __restrict__ o, int E) {
  int i = blockIdx.x * blockDim.x + threadIdx.x;
  if (i < E) {
    const int r = row[i], c = col[i];
    o[(size_t)i * 4 + 0] = ea[i];
    o[(size_t)i * 4 + 1] = x[(size_t)r * 8 + 0] - x[(size_t)c * 8 + 0];
    o[(size_t)i * 4 + 2] = x[(size_t)r * 8 + 3] - x[(size_t)c * 8 + 3];
    o[(size_t)i * 4 + 3] = 0.f;
  }
}

__global__ void k_starts(const int* __restrict__ batch, int* __restrict__ starts,
                         int N) {
  int i = blockIdx.x * blockDim.x + threadIdx.x;
  if (i < N) {
    if (i == 0) { starts[0] = 0; starts[NUM_GRAPHS] = N; }
    else if (batch[i] != batch[i - 1]) starts[batch[i]] = i;
  }
}

// per-graph sum / min / max / variance over sorted node ranges
__global__ void k_stats(const float* __restrict__ h, const int* __restrict__ starts,
                        float* __restrict__ s, float* __restrict__ mi,
                        float* __restrict__ ma, float* __restrict__ sq) {
  __shared__ float rs[4][64], rq[4][64], rn[4][64], rx[4][64];
  const int g = blockIdx.x;
  const int st = starts[g], en = starts[g + 1];
  const int t = threadIdx.x, f = t & 63, grp = t >> 6;
  float sum = 0.f, ssq = 0.f, mn = 3.4e38f, mx = -3.4e38f;
  for (int i = st + grp; i < en; i += 4) {
    const float v = h[(size_t)i * 64 + f];
    sum += v; ssq += v * v; mn = fminf(mn, v); mx = fmaxf(mx, v);
  }
  rs[grp][f] = sum; rq[grp][f] = ssq; rn[grp][f] = mn; rx[grp][f] = mx;
  __syncthreads();
  if (grp == 0) {
    for (int k = 1; k < 4; ++k) {
      sum += rs[k][f]; ssq += rq[k][f];
      mn = fminf(mn, rn[k][f]); mx = fmaxf(mx, rx[k][f]);
    }
    const float c  = (float)(en - st);
    const float me = sum / c;
    s[g * 64 + f]  = sum;
    mi[g * 64 + f] = mn;
    ma[g * 64 + f] = mx;
    sq[g * 64 + f] = ssq / c - me * me;
  }
}

// decoder: per graph, 4 heads of (LN -> 64x64 -> relu -> LN -> 64x1 -> relu)
__global__ void k_decoder(const float* __restrict__ gm, DecHead h0, DecHead h1,
                          DecHead h2, DecHead h3, float* __restrict__ out) {
  __shared__ float red[64];
  __shared__ float xn[64];
  const int r = blockIdx.x, t = threadIdx.x;   // 64 threads
  DecHead hs[4] = {h0, h1, h2, h3};
  const float x0 = gm[r * 64 + t];
  #pragma unroll
  for (int hd = 0; hd < 4; ++hd) {
    const DecHead Hh = hs[hd];
    red[t] = x0; __syncthreads();
    for (int o = 32; o > 0; o >>= 1) { if (t < o) red[t] += red[t + o]; __syncthreads(); }
    const float m1 = red[0] * (1.f / 64.f); __syncthreads();
    const float d1 = x0 - m1;
    red[t] = d1 * d1; __syncthreads();
    for (int o = 32; o > 0; o >>= 1) { if (t < o) red[t] += red[t + o]; __syncthreads(); }
    const float rs1 = rsqrtf(red[0] * (1.f / 64.f) + 1e-5f); __syncthreads();
    xn[t] = d1 * rs1 * Hh.g0[t] + Hh.b0[t]; __syncthreads();
    float y = Hh.bb0[t];
    for (int k = 0; k < 64; ++k) y += xn[k] * Hh.W0[k * 64 + t];
    y = fmaxf(y, 0.f);
    __syncthreads();
    red[t] = y; __syncthreads();
    for (int o = 32; o > 0; o >>= 1) { if (t < o) red[t] += red[t + o]; __syncthreads(); }
    const float m2 = red[0] * (1.f / 64.f); __syncthreads();
    const float d2 = y - m2;
    red[t] = d2 * d2; __syncthreads();
    for (int o = 32; o > 0; o >>= 1) { if (t < o) red[t] += red[t + o]; __syncthreads(); }
    const float rs2 = rsqrtf(red[0] * (1.f / 64.f) + 1e-5f); __syncthreads();
    const float z = d2 * rs2 * Hh.g1[t] + Hh.b1[t];
    red[t] = z * Hh.W1[t]; __syncthreads();
    for (int o = 32; o > 0; o >>= 1) { if (t < o) red[t] += red[t + o]; __syncthreads(); }
    if (t == 0) out[r * 4 + hd] = fmaxf(red[0] + Hh.bb1[0], 0.f);
    __syncthreads();
  }
}

// ------------------------------- host side ---------------------------------

struct MLPF { const float *W0,*b0,*W1,*b1,*W2,*b2,*g,*bb,*W3,*b3; };

static inline size_t mlp_shmem(int kin) {
  return (size_t)4 * (16 * (kin + 4) + 16 * 68) * sizeof(float);
}

extern "C" void kernel_launch(void* const* d_in, const int* in_sizes, int n_in,
                              void* d_out, int out_size, void* d_ws, size_t ws_size,
                              hipStream_t stream) {
  (void)out_size; (void)ws_size;
  // locate the four data tensors by (unique) element counts
  const float* x = nullptr; const float* ea = nullptr;
  const int* ei = nullptr;  const int* batch = nullptr;
  for (int i = 0; i < n_in; ++i) {
    switch (in_sizes[i]) {
      case N_NODES * 8: x     = (const float*)d_in[i]; break;
      case N_EDGES:     ea    = (const float*)d_in[i]; break;
      case 2 * N_EDGES: ei    = (const int*)d_in[i];   break;
      case N_NODES:     batch = (const int*)d_in[i];   break;
      default: break;
    }
  }
  const int* row = ei;
  const int* col = ei + N_EDGES;

  // params leaf base: either dict-insertion order (params first) or whole-dict
  // sorted flatten (batch, edge_attr, edge_index first).
  const int pBase = (in_sizes[0] == N_NODES) ? 3 : 0;
  // within params: JAX sorted-key pytree (first leaf = conv0.edge.b, 64 elems)
  // vs plain insertion order (first leaf = node_enc.lin0.W, 512 elems).
  const bool sortedKeys = (in_sizes[pBase] == 64);

  auto L = [&](int k) { return (const float*)d_in[pBase + k]; };
  auto mkMLP = [&](int o) {
    MLPF w;
    if (sortedKeys) {  // b, g, W0,b0, W1,b1, W2,b2, W3,b3
      w.bb = L(o);     w.g = L(o + 1);
      w.W0 = L(o + 2); w.b0 = L(o + 3);
      w.W1 = L(o + 4); w.b1 = L(o + 5);
      w.W2 = L(o + 6); w.b2 = L(o + 7);
      w.W3 = L(o + 8); w.b3 = L(o + 9);
    } else {           // W0,b0, W1,b1, W2,b2, g, b, W3,b3
      w.W0 = L(o);     w.b0 = L(o + 1);
      w.W1 = L(o + 2); w.b1 = L(o + 3);
      w.W2 = L(o + 4); w.b2 = L(o + 5);
      w.g  = L(o + 6); w.bb = L(o + 7);
      w.W3 = L(o + 8); w.b3 = L(o + 9);
    }
    return w;
  };

  MLPF f_ne = mkMLP(sortedKeys ? 162 : 0);
  MLPF f_ee = mkMLP(sortedKeys ? 152 : 10);
  MLPF f_edge[3], f_node1[3], f_node2[3], f_glob[3];
  for (int ci = 0; ci < 3; ++ci) {
    if (sortedKeys) {  // conv keys sorted: edge, glob, node1, node2
      const int b = 40 * ci;
      f_edge[ci]  = mkMLP(b);
      f_glob[ci]  = mkMLP(b + 10);
      f_node1[ci] = mkMLP(b + 20);
      f_node2[ci] = mkMLP(b + 30);
    } else {           // insertion: edge, node1, node2, glob
      const int b = 20 + 40 * ci;
      f_edge[ci]  = mkMLP(b);
      f_node1[ci] = mkMLP(b + 10);
      f_node2[ci] = mkMLP(b + 20);
      f_glob[ci]  = mkMLP(b + 30);
    }
  }
  const int decBase = sortedKeys ? 120 : 140;
  DecHead heads[4];
  for (int i = 0; i < 4; ++i) {
    const int b = decBase + 8 * i;
    heads[i] = {L(b), L(b + 1), L(b + 2), L(b + 3),
                L(b + 4), L(b + 5), L(b + 6), L(b + 7)};
  }

  // workspace carve
  char* wp = (char*)d_ws;
  auto carve = [&](size_t bytes) {
    void* r = (void*)wp;
    wp += (bytes + 255) & ~(size_t)255;
    return r;
  };
  float* h    = (float*)carve((size_t)N_NODES * 64 * 4);
  float* e    = (float*)carve((size_t)N_EDGES * 64 * 4);
  float* agg  = (float*)carve((size_t)N_NODES * 64 * 4);
  float* u    = (float*)carve((size_t)NUM_GRAPHS * 64 * 4);
  float* sbuf = (float*)carve((size_t)NUM_GRAPHS * 64 * 4);
  float* mib  = (float*)carve((size_t)NUM_GRAPHS * 64 * 4);
  float* mab  = (float*)carve((size_t)NUM_GRAPHS * 64 * 4);
  float* sqb  = (float*)carve((size_t)NUM_GRAPHS * 64 * 4);
  float* eenc = (float*)carve((size_t)N_EDGES * 4 * 4);
  int*   eb   = (int*)carve((size_t)N_EDGES * 4);
  int*   starts = (int*)carve((size_t)(NUM_GRAPHS + 1) * 4);

  // build packed-weight table (bf16, B-frag layout)
  RepackTab tab; int tn = 0;
  auto addW = [&](const float* src, int K, int Ksrc) -> const __bf16* {
    __bf16* dst = (__bf16*)carve((size_t)K * 64 * 2);
    tab.e[tn++] = {src, dst, K, Ksrc};
    return dst;
  };
  auto packMLP = [&](const MLPF& f, int K0, int K0src) {
    MLPW w;
    w.W0 = addW(f.W0, K0, K0src);
    w.W1 = addW(f.W1, 64, 64);
    w.W2 = addW(f.W2, 64, 64);
    w.W3 = addW(f.W3, 64, 64);
    w.b0 = f.b0; w.b1 = f.b1; w.b2 = f.b2;
    w.g = f.g; w.bb = f.bb; w.b3 = f.b3;
    return w;
  };
  MLPW ne = packMLP(f_ne, 32, 8);    // node_enc W0 padded 8 -> 32
  MLPW ee = packMLP(f_ee, 32, 3);    // edge_enc W0 padded 3 -> 32
  MLPW cedge[3], cnode1[3], cnode2[3], cglob[3];
  for (int ci = 0; ci < 3; ++ci) {
    cedge[ci]  = packMLP(f_edge[ci], 256, 256);
    cnode1[ci] = packMLP(f_node1[ci], 128, 128);
    cnode2[ci] = packMLP(f_node2[ci], 192, 192);
    cglob[ci]  = packMLP(f_glob[ci], 320, 320);
  }

  const int gE = (N_EDGES + 255) / 256;
  const int gN = (N_NODES + 255) / 256;

  k_repack<<<tn, 256, 0, stream>>>(tab);
  k_eb    <<<gE, 256, 0, stream>>>(row, batch, eb, N_EDGES);
  k_eenc  <<<gE, 256, 0, stream>>>(x, ea, row, col, eenc, N_EDGES);
  k_starts<<<gN, 256, 0, stream>>>(batch, starts, N_NODES);
  k_zero  <<<(NUM_GRAPHS * 64 + 255) / 256, 256, 0, stream>>>(u, NUM_GRAPHS * 64);

  const Seg Z = {nullptr, nullptr, 0};
  const int ntE = N_EDGES / 16, ntN = N_NODES / 16, ntG = NUM_GRAPHS / 16;

  {  // h = mlp(node_enc, x)          (K padded 8 -> 32)
    Seg a = {x, nullptr, 8};
    fused_mlp<32><<<(ntN + 3) / 4, 128, mlp_shmem(32), stream>>>(
        ne, a, Z, Z, Z, Z, 1, h, nullptr, OUT_STORE, ntN);
  }
  {  // e = mlp(edge_enc, e_enc)      (K padded 4 -> 32)
    Seg a = {eenc, nullptr, 4};
    fused_mlp<32><<<(ntE + 3) / 4, 128, mlp_shmem(32), stream>>>(
        ee, a, Z, Z, Z, Z, 1, e, nullptr, OUT_STORE, ntE);
  }

  for (int ci = 0; ci < 3; ++ci) {
    {  // e += mlp(edge, [h[row], h[col], e, u[eb]])
      Seg a = {h, row, 64}, b = {h, col, 64}, c = {e, nullptr, 64}, d = {u, eb, 64};
      fused_mlp<256><<<(ntE + 3) / 4, 128, mlp_shmem(256), stream>>>(
          cedge[ci], a, b, c, d, Z, 4, e, nullptr, OUT_ADD, ntE);
    }
    k_zero<<<((N_NODES * 64) + 255) / 256, 256, 0, stream>>>(agg, (long long)N_NODES * 64);
    {  // agg = segment_sum(mlp(node1, [h[row], e]), col)
      Seg a = {h, row, 64}, b = {e, nullptr, 64};
      fused_mlp<128><<<(ntE + 3) / 4, 128, mlp_shmem(128), stream>>>(
          cnode1[ci], a, b, Z, Z, Z, 2, agg, col, OUT_SCATTER, ntE);
    }
    {  // h += mlp(node2, [h, agg, u[batch]])
      Seg a = {h, nullptr, 64}, b = {agg, nullptr, 64}, c = {u, batch, 64};
      fused_mlp<192><<<(ntN + 3) / 4, 128, mlp_shmem(192), stream>>>(
          cnode2[ci], a, b, c, Z, Z, 3, h, nullptr, OUT_ADD, ntN);
    }
    k_stats<<<NUM_GRAPHS, 256, 0, stream>>>(h, starts, sbuf, mib, mab, sqb);
    {  // u += mlp(glob, [u, s, mi, ma, var])
      Seg a = {u, nullptr, 64}, b = {sbuf, nullptr, 64}, c = {mib, nullptr, 64},
          d = {mab, nullptr, 64}, f = {sqb, nullptr, 64};
      fused_mlp<320><<<(ntG + 3) / 4, 128, mlp_shmem(320), stream>>>(
          cglob[ci], a, b, c, d, f, 5, u, nullptr, OUT_ADD, ntG);
    }
  }

  // final readout g == segment_sum(h, batch) == sbuf from the last conv's stats
  k_decoder<<<NUM_GRAPHS, 64, 0, stream>>>(sbuf, heads[0], heads[1], heads[2],
                                           heads[3], (float*)d_out);
}
